// MultiHeadAttention_51505247813938
// MI455X (gfx1250) — matrix operand order repaired and checked by reading, not compiled, not run on hardware
//
#include <hip/hip_runtime.h>

// ---------------------------------------------------------------------------
// MI455X (gfx1250) multi-head attention:
//   Q/K/V projections + flash attention + output projection via
//   v_wmma_f32_16x16x32_bf16 (wave32 WMMA, f32 accumulate).
//   K tiles staged LDS-side by the Tensor Data Mover (tensor_load_to_lds).
// ---------------------------------------------------------------------------

typedef __attribute__((ext_vector_type(16))) __bf16    v16bf;
typedef __attribute__((ext_vector_type(8)))  __bf16    v8bf;
typedef __attribute__((ext_vector_type(8)))  float     v8f;
typedef __attribute__((ext_vector_type(4)))  unsigned  v4u;
typedef __attribute__((ext_vector_type(8)))  unsigned  v8u;

#define DEV __device__ __forceinline__

DEV v8f wmma_bf16(v16bf a, v16bf b, v8f c) {
  // (neg_a, A, neg_b, B, c_mod, C, reuse_a, reuse_b)
  return __builtin_amdgcn_wmma_f32_16x16x32_bf16(false, a, false, b, (short)0, c,
                                                 false, false);
}

DEV v16bf mk16(v8bf lo, v8bf hi) {
  v16bf r;
#pragma unroll
  for (int i = 0; i < 8; ++i) { r[i] = lo[i]; r[i + 8] = hi[i]; }
  return r;
}

// A-fragment 16x32 bf16 (ISA 7.12.2): per lane row = caller's row,
// elements 0..7 -> k = kc + 8*half + j ; elements 8..15 -> k = kc + 16 + 8*half + j
DEV v16bf load_a_frag(const __bf16* base, int ld, int row, int kc, int half) {
  const __bf16* p = base + (size_t)row * ld + kc + 8 * half;
  return mk16(*(const v8bf*)p, *(const v8bf*)(p + 16));
}

// B-fragment 32x16 bf16 read from N-major storage [n][k] (leading dim ld):
// per lane column n, elements 0..7 -> k = kc + 8*half + j ; elements 8..15 -> k = kc + 16 + 8*half + j
DEV v16bf load_b_frag(const __bf16* baseN, int ld, int n, int kc, int half) {
  const __bf16* p = baseN + (size_t)n * ld + kc + 8 * half;
  return mk16(*(const v8bf*)p, *(const v8bf*)(p + 16));
}

// TDM: DMA a 2D bf16 tile [rows x 64] from a row-major [tensorRows x 64]
// tensor into LDS. D# per CDNA5 ISA.
// Caller must be a single wave (scalar-guarded); completion via TENSORcnt.
DEV void tdm_load_tile_2d_bf16(unsigned ldsAddr, const __bf16* gsrc,
                               unsigned rows, unsigned cols,
                               unsigned tensorRows) {
  const unsigned long long ga = (unsigned long long)(size_t)gsrc;
  v4u g0;
  g0[0] = 1u;                                   // count=1, user descriptor
  g0[1] = ldsAddr;                              // lds_addr (bytes)
  g0[2] = (unsigned)ga;                         // global_addr[31:0]
  g0[3] = ((unsigned)(ga >> 32) & 0x01FFFFFFu)  // global_addr[56:32]
          | (2u << 30);                         // type=2 ("image")
  v8u g1;
  g1[0] = 1u << 16;                 // wg_mask=0, data_size=1 (2 bytes)
  g1[1] = (cols & 0xFFFFu) << 16;   // tensor_dim0[15:0]  (row length)
  g1[2] = (tensorRows & 0xFFFFu) << 16;  // tensor_dim1[15:0]
  g1[3] = (cols & 0xFFFFu) << 16;   // tile_dim0
  g1[4] = rows & 0xFFFFu;           // tile_dim1 (tile_dim2=0 -> 2D)
  g1[5] = cols;                     // tensor_dim0_stride[31:0]
  g1[6] = 0u;                       // stride0[47:32]=0, dim1_stride lo=0
  g1[7] = 0u;
  asm volatile("tensor_load_to_lds %0, %1" ::"s"(g0), "s"(g1) : "memory");
}

// ---------------------------------------------------------------------------
// fp32 -> bf16 conversion (weights), 4 elements/thread, n fixed = 1024*1024
// ---------------------------------------------------------------------------
__global__ __launch_bounds__(256) void cvt_f32_to_bf16_x4(
    const float* __restrict__ in, __bf16* __restrict__ out) {
  const int i = (blockIdx.x * 256 + threadIdx.x) * 4;
  const float4 v = *(const float4*)(in + i);
  out[i + 0] = (__bf16)v.x;
  out[i + 1] = (__bf16)v.y;
  out[i + 2] = (__bf16)v.z;
  out[i + 3] = (__bf16)v.w;
}

// ---------------------------------------------------------------------------
// GEMM: C[8192x1024] = A[8192x1024] * W[1024x1024] + bias
// block = 256 threads (8 waves), 128x128 tile, k-step 32.
// Wave grid 4x2: each wave computes 32x64 via 2x4 wmma accumulators.
// EPI==0: scatter bf16 into [B=4,H=16,S=2048,Dh=64] (head-split) layout.
// EPI==1: fp32 row-major output.
// ---------------------------------------------------------------------------
template <typename AT, int EPI>
__global__ __launch_bounds__(256) void gemm128x128(
    const AT* __restrict__ A, const __bf16* __restrict__ W,
    const float* __restrict__ bias, void* __restrict__ outp) {
  constexpr int K = 1024, N = 1024;
  __shared__ __align__(16) __bf16 Asm[128 * 32];  // [m][k]
  __shared__ __align__(16) __bf16 Bsm[128 * 32];  // transposed: [n][k]
  const int tid = threadIdx.x;
  const int lane = tid & 31, wave = tid >> 5;
  const int half = lane >> 4, l16 = lane & 15;
  const int wm = wave & 3, wn = wave >> 2;
  const int blockM = blockIdx.y * 128, blockN = blockIdx.x * 128;

  v8f acc[2][4];
#pragma unroll
  for (int i = 0; i < 2; ++i)
#pragma unroll
    for (int j = 0; j < 4; ++j)
#pragma unroll
      for (int r = 0; r < 8; ++r) acc[i][j][r] = 0.f;

  for (int k0 = 0; k0 < K; k0 += 32) {
    __syncthreads();
    {  // stage A tile (convert to bf16 on the way in)
      const int row = tid >> 1, kk = (tid & 1) * 16;
      const AT* ap = A + (size_t)(blockM + row) * K + k0 + kk;
      __bf16* dst = Asm + row * 32 + kk;
#pragma unroll
      for (int j = 0; j < 16; ++j) dst[j] = (__bf16)(float)ap[j];
    }
    {  // stage W tile transposed to N-major
      const int kk = tid >> 3, n0 = (tid & 7) * 16;
      const __bf16* wp = W + (size_t)(k0 + kk) * N + blockN + n0;
#pragma unroll
      for (int j = 0; j < 16; ++j) Bsm[(n0 + j) * 32 + kk] = wp[j];
    }
    __syncthreads();

    v16bf af[2], bfr[4];
#pragma unroll
    for (int i = 0; i < 2; ++i)
      af[i] = load_a_frag(Asm, 32, wm * 32 + i * 16 + l16, 0, half);
#pragma unroll
    for (int j = 0; j < 4; ++j)
      bfr[j] = load_b_frag(Bsm, 32, wn * 64 + j * 16 + l16, 0, half);
#pragma unroll
    for (int i = 0; i < 2; ++i)
#pragma unroll
      for (int j = 0; j < 4; ++j) acc[i][j] = wmma_bf16(af[i], bfr[j], acc[i][j]);
  }

#pragma unroll
  for (int i = 0; i < 2; ++i)
#pragma unroll
    for (int j = 0; j < 4; ++j)
#pragma unroll
      for (int r = 0; r < 8; ++r) {
        const int m = blockM + wm * 32 + i * 16 + r + 8 * half;
        const int n = blockN + wn * 64 + j * 16 + l16;
        const float v = acc[i][j][r] + bias[n];
        if (EPI == 0) {
          // head-split scatter: b = m/2048, s = m%2048, h = n/64, dh = n%64
          __bf16* o = (__bf16*)outp;
          o[(((size_t)(m >> 11) * 16 + (n >> 6)) * 2048 + (m & 2047)) * 64 +
            (n & 63)] = (__bf16)v;
        } else {
          ((float*)outp)[(size_t)m * 1024 + n] = v;
        }
      }
}

// ---------------------------------------------------------------------------
// Flash attention: grid (S/128, H, B), block 256 (8 waves).
// Each wave owns 16 q-rows; k-loop over 64-key tiles.
// K tile DMA'd into LDS by the TDM as [key][dh] (== N-major for QK^T B-frags),
// V tile in LDS transposed [dh][key] (== N-major for PV B-frags),
// P re-laid out per-wave through LDS (C-frag -> A-frag).
// ---------------------------------------------------------------------------
__global__ __launch_bounds__(256) void flash_attn(
    const __bf16* __restrict__ Qg, const __bf16* __restrict__ Kg,
    const __bf16* __restrict__ Vg, const int* __restrict__ mask,
    __bf16* __restrict__ Og) {
  constexpr int S = 2048, DH = 64, H = 16;
  __shared__ __align__(16) __bf16 Ksm[64 * 64];       // [key][dh]   (TDM dest)
  __shared__ __align__(16) __bf16 Vtsm[64 * 64];      // [dh][key]
  __shared__ __align__(16) __bf16 Psm[8 * 16 * 64];   // per-wave [row][key]
  const int tid = threadIdx.x;
  const int lane = tid & 31, wave = tid >> 5;
  const int half = lane >> 4, l16 = lane & 15;
  const int h = blockIdx.y, b = blockIdx.z;
  const size_t headoff = ((size_t)b * H + h) * (size_t)S * DH;
  const int qrow0 = blockIdx.x * 128 + wave * 16;
  // scalar wave id -> guaranteed s_cbranch guard for the TDM issue
  const int wave_s = __builtin_amdgcn_readfirstlane(tid) >> 5;
  const unsigned ldsK = (unsigned)(size_t)(&Ksm[0]);  // LDS byte address

  // Q fragments held in registers for the whole kernel (Dh = 64 -> 2 chunks)
  v16bf qa[2];
#pragma unroll
  for (int c = 0; c < 2; ++c)
    qa[c] = load_a_frag(Qg + headoff, DH, qrow0 + l16, 32 * c, half);

  v8f o[4];
  float mrow[8], lrow[8];
#pragma unroll
  for (int d = 0; d < 4; ++d)
#pragma unroll
    for (int r = 0; r < 8; ++r) o[d][r] = 0.f;
#pragma unroll
  for (int r = 0; r < 8; ++r) { mrow[r] = -1e30f; lrow[r] = 0.f; }

  for (int kb = 0; kb < S; kb += 64) {
    __syncthreads();  // previous iteration's LDS reads complete
    if (wave_s == 0) {  // one TDM issue per block: K tile 64x64 bf16 -> LDS
      tdm_load_tile_2d_bf16(ldsK, Kg + headoff + (size_t)kb * DH,
                            /*rows=*/64, /*cols=*/DH, /*tensorRows=*/S);
    }
#pragma unroll
    for (int it = 0; it < 2; ++it) {  // V tile: transpose into LDS (VALU path)
      const int idx = tid + it * 256;
      const int row = idx >> 3, cb = (idx & 7) * 8;
      const v8bf v =
          *(const v8bf*)(Vg + headoff + (size_t)(kb + row) * DH + cb);
#pragma unroll
      for (int j = 0; j < 8; ++j) Vtsm[(cb + j) * 64 + row] = v[j];
    }
    if (wave_s == 0) __builtin_amdgcn_s_wait_tensorcnt(0);  // K tile landed
    __syncthreads();

    // scores: s[t] = Q(16x64) * K_tile^T cols [16t,16t+16)
    v8f s[4];
#pragma unroll
    for (int t = 0; t < 4; ++t) {
#pragma unroll
      for (int r = 0; r < 8; ++r) s[t][r] = 0.f;
      const v16bf b0 = load_b_frag(Ksm, 64, 16 * t + l16, 0, half);
      const v16bf b1 = load_b_frag(Ksm, 64, 16 * t + l16, 32, half);
      s[t] = wmma_bf16(qa[0], b0, s[t]);
      s[t] = wmma_bf16(qa[1], b1, s[t]);
    }

    // scale 1/sqrt(64) and mask (mask==0 -> -1e9), as in the reference
#pragma unroll
    for (int t = 0; t < 4; ++t)
#pragma unroll
      for (int r = 0; r < 8; ++r) {
        const int kcol = kb + 16 * t + l16;
        const int qr = qrow0 + r + 8 * half;
        const int mv = mask[((size_t)b * S + qr) * S + kcol];
        const float sv = s[t][r] * 0.125f;
        s[t][r] = (mv == 0) ? -1e9f : sv;
      }

    // online softmax, per-row (rows live inside one 16-lane half)
    float alpha[8];
#pragma unroll
    for (int r = 0; r < 8; ++r) {
      float mx = fmaxf(fmaxf(s[0][r], s[1][r]), fmaxf(s[2][r], s[3][r]));
#pragma unroll
      for (int off = 8; off >= 1; off >>= 1)
        mx = fmaxf(mx, __shfl_xor(mx, off, 32));
      const float mnew = fmaxf(mrow[r], mx);
      alpha[r] = __expf(mrow[r] - mnew);
      float sum = 0.f;
#pragma unroll
      for (int t = 0; t < 4; ++t) {
        const float p = __expf(s[t][r] - mnew);
        s[t][r] = p;
        sum += p;
      }
#pragma unroll
      for (int off = 8; off >= 1; off >>= 1) sum += __shfl_xor(sum, off, 32);
      lrow[r] = lrow[r] * alpha[r] + sum;
      mrow[r] = mnew;
    }
#pragma unroll
    for (int d = 0; d < 4; ++d)
#pragma unroll
      for (int r = 0; r < 8; ++r) o[d][r] *= alpha[r];

    // P: C-layout -> per-wave LDS -> A-fragments (no cross-wave hazard)
    __bf16* Pw = Psm + wave * 16 * 64;
#pragma unroll
    for (int t = 0; t < 4; ++t)
#pragma unroll
      for (int r = 0; r < 8; ++r)
        Pw[(r + 8 * half) * 64 + 16 * t + l16] = (__bf16)s[t][r];

    // O += P(16x64) * V_tile(64x64)
#pragma unroll
    for (int c = 0; c < 2; ++c) {
      const v16bf pa = load_a_frag(Pw, 64, l16, 32 * c, half);
#pragma unroll
      for (int d = 0; d < 4; ++d) {
        const v16bf vb = load_b_frag(Vtsm, 64, 16 * d + l16, 32 * c, half);
        o[d] = wmma_bf16(pa, vb, o[d]);
      }
    }
  }

  // normalize and write attn output bf16 [B, S, H*Dh]
#pragma unroll
  for (int r = 0; r < 8; ++r) {
    const float inv = 1.0f / lrow[r];
    const int qr = qrow0 + r + 8 * half;
#pragma unroll
    for (int d = 0; d < 4; ++d) {
      const int col = h * 64 + 16 * d + l16;
      Og[((size_t)b * S + qr) * 1024 + col] = (__bf16)(o[d][r] * inv);
    }
  }
}

// ---------------------------------------------------------------------------
// Launch
// ---------------------------------------------------------------------------
extern "C" void kernel_launch(void* const* d_in, const int* in_sizes, int n_in,
                              void* d_out, int out_size, void* d_ws,
                              size_t ws_size, hipStream_t stream) {
  (void)in_sizes; (void)n_in; (void)out_size; (void)ws_size;
  const float* query = (const float*)d_in[0];
  const float* key_  = (const float*)d_in[1];
  const float* value = (const float*)d_in[2];
  const int*   mask  = (const int*)d_in[3];
  const float* Wq = (const float*)d_in[4];  const float* bq = (const float*)d_in[5];
  const float* Wk = (const float*)d_in[6];  const float* bk = (const float*)d_in[7];
  const float* Wv = (const float*)d_in[8];  const float* bv = (const float*)d_in[9];
  const float* Wo = (const float*)d_in[10]; const float* bo = (const float*)d_in[11];

  char* ws = (char*)d_ws;
  const size_t WB = 1024ull * 1024 * 2;          // one bf16 weight matrix
  const size_t TB = 4ull * 16 * 2048 * 64 * 2;   // one bf16 [B,H,S,Dh] tensor
  __bf16* wWq = (__bf16*)(ws);
  __bf16* wWk = (__bf16*)(ws + WB);
  __bf16* wWv = (__bf16*)(ws + 2 * WB);
  __bf16* wWo = (__bf16*)(ws + 3 * WB);
  __bf16* Qb   = (__bf16*)(ws + 4 * WB);
  __bf16* Kb   = (__bf16*)(ws + 4 * WB + TB);
  __bf16* Vb   = (__bf16*)(ws + 4 * WB + 2 * TB);
  __bf16* Attn = (__bf16*)(ws + 4 * WB + 3 * TB);

  const int cvtBlocks = (1024 * 1024) / (256 * 4);
  cvt_f32_to_bf16_x4<<<cvtBlocks, 256, 0, stream>>>(Wq, wWq);
  cvt_f32_to_bf16_x4<<<cvtBlocks, 256, 0, stream>>>(Wk, wWk);
  cvt_f32_to_bf16_x4<<<cvtBlocks, 256, 0, stream>>>(Wv, wWv);
  cvt_f32_to_bf16_x4<<<cvtBlocks, 256, 0, stream>>>(Wo, wWo);

  dim3 gG(1024 / 128, 8192 / 128);  // (N tiles, M tiles)
  gemm128x128<float, 0><<<gG, 256, 0, stream>>>(query, wWq, bq, (void*)Qb);
  gemm128x128<float, 0><<<gG, 256, 0, stream>>>(key_,  wWk, bk, (void*)Kb);
  gemm128x128<float, 0><<<gG, 256, 0, stream>>>(value, wWv, bv, (void*)Vb);

  dim3 gF(2048 / 128, 16, 4);  // (q tiles, H, B)
  flash_attn<<<gF, 256, 0, stream>>>(Qb, Kb, Vb, mask, Attn);

  gemm128x128<__bf16, 1><<<gG, 256, 0, stream>>>(Attn, wWo, bo, d_out);
}
